// MemNet_84559316124212
// MI455X (gfx1250) — matrix operand order repaired and checked by reading, not compiled, not run on hardware
//
#include <hip/hip_runtime.h>
#include <math.h>

// ---------------------------------------------------------------------------
// MemNet on MI455X (gfx1250, wave32, WMMA).
// Dominant cost: kx = (w * emb[text]) @ Wk^T + bk  (262144x300 @ 300x300)
// -> fused gather + f16 WMMA GEMM (f32 accum); kx materialized once (319 MB)
//    then streamed 4x (k_score + 3 hops) => HBM-bound ~1.6 GB @ 23.3 TB/s.
// Wq-hop GEMM eliminated via u = Wq^T @ wq_part precompute.
// Wk/Wx/Wp pre-converted once to zero-padded f16 [304][320] so B fragments
// are single 32B loads; A tile staged in LDS as f16 with double buffering.
// ---------------------------------------------------------------------------

#define BB   512
#define SS   512
#define DD   300
#define PP   3
#define LDKX 304      // kx row stride (300 padded to 19 n-tiles * 16)
#define NTILES 19     // ceil(300/16)
#define KCHUNKS 10    // 10 * 32 = 320 >= 300 (zero padded)
#define WH_LDN 320    // f16 weight row stride (K padded)
#define WH_ROWS 304   // f16 weight rows (N padded)

typedef __attribute__((ext_vector_type(16))) _Float16 v16h;
typedef __attribute__((ext_vector_type(8)))  _Float16 v8h;
typedef __attribute__((ext_vector_type(8)))  float    v8f;

// ---------------------------------------------------------------------------
// W [300x300] f32 -> Wh [304][320] f16, zero padded
// ---------------------------------------------------------------------------
__global__ __launch_bounds__(256)
void convert_w_kernel(const float* __restrict__ W, _Float16* __restrict__ Wh)
{
    const int idx = blockIdx.x * 256 + threadIdx.x;
    if (idx >= WH_ROWS * WH_LDN) return;
    const int n = idx / WH_LDN, k = idx % WH_LDN;
    const float v = (n < DD && k < DD) ? W[n * DD + k] : 0.f;
    Wh[idx] = (_Float16)v;
}

// ---------------------------------------------------------------------------
// C[m,n] = sum_k A[m,k]*W[n,k] + bias[n] (+ res[m,n]) via v_wmma_f32_16x16x32_f16
// A source: tokens!=null -> gather emb[tokens[m]]*wscale[m]; else Asrc[m*300+k].
// Block: 256 thr = 8 waves; 64-row M tile; wave w owns n-tiles {w, w+8, w+16}.
// ---------------------------------------------------------------------------
__global__ __launch_bounds__(256)
void wmma_gemm_nt(const float* __restrict__ Asrc,
                  const int*   __restrict__ tokens,
                  const float* __restrict__ emb,
                  const float* __restrict__ wscale,
                  const _Float16* __restrict__ Wh,   // [304][320] f16 padded
                  const float* __restrict__ bias,    // [300] or null
                  const float* __restrict__ res,     // [M x 300] or null
                  float* __restrict__ C, int ldc,
                  int M, int Nstore)
{
    __shared__ __align__(16) _Float16 As[2][64][40];  // 2 x 64x32 tile, stride 40 halves

    const int tid  = threadIdx.x;
    const int mb   = blockIdx.x * 64;
    const int wave = tid >> 5;
    const int lane = tid & 31;
    const int lm   = lane & 15;               // A:M / B:N / C:N index
    const int lh   = lane >> 4;               // half select

    // ---- per-thread A staging setup (hoisted out of K loop) ----
    const int  srow = tid >> 2;               // 64 rows, 4 threads/row
    const int  sc0  = (tid & 3) * 8;          // 8 consecutive k each
    const float* src;
    float sc;
    if (tokens) {
        const int tok = tokens[mb + srow];
        src = emb + (long)tok * DD;           // emb[0] == 0 -> no tok guard needed
        sc  = wscale[mb + srow];
    } else {
        src = Asrc + (long)(mb + srow) * DD;
        sc  = 1.f;
    }

    auto stage = [&](int kc) {
        const int kb = kc * 32;
        v8h h;
        if (kc < KCHUNKS - 1) {               // fully in-range: 2x float4 loads
            const float4 a = *(const float4*)(src + kb + sc0);
            const float4 b = *(const float4*)(src + kb + sc0 + 4);
            h[0] = (_Float16)(a.x * sc); h[1] = (_Float16)(a.y * sc);
            h[2] = (_Float16)(a.z * sc); h[3] = (_Float16)(a.w * sc);
            h[4] = (_Float16)(b.x * sc); h[5] = (_Float16)(b.y * sc);
            h[6] = (_Float16)(b.z * sc); h[7] = (_Float16)(b.w * sc);
        } else {                              // tail chunk: clamp k, zero pad
            #pragma unroll
            for (int i = 0; i < 8; ++i) {
                const int k   = kb + sc0 + i;
                const int kcl = (k < DD) ? k : 0;
                const float v = src[kcl] * sc;
                h[i] = (_Float16)((k < DD) ? v : 0.f);
            }
        }
        *(v8h*)&As[kc & 1][srow][sc0] = h;    // one ds_store_b128
    };

    v8f acc[4][3];
    #pragma unroll
    for (int a = 0; a < 4; ++a)
        #pragma unroll
        for (int t = 0; t < 3; ++t)
            #pragma unroll
            for (int i = 0; i < 8; ++i) acc[a][t][i] = 0.f;

    // B row pointers (uniform per lane within wave; rows always < 304 pad rows)
    const _Float16* wrow[3];
    #pragma unroll
    for (int t = 0; t < 3; ++t) {
        const int nt = wave + t * 8;
        wrow[t] = Wh + (size_t)((nt < NTILES ? nt : 0) * 16 + lm) * WH_LDN + 16 * lh;
    }

    stage(0);
    for (int kc = 0; kc < KCHUNKS; ++kc) {
        __syncthreads();                      // single barrier per K step
        if (kc + 1 < KCHUNKS) stage(kc + 1);  // fill other LDS buffer

        const int kb = kc * 32;
        const int p  = kc & 1;

        // A fragments: two contiguous 16B spans per lane (contiguous k = 16*lh + s, same order as B)
        v16h afr[4];
        #pragma unroll
        for (int mt = 0; mt < 4; ++mt) {
            const int r = mt * 16 + lm;
            union { v16h v; v8h h[2]; } u;
            u.h[0] = *(const v8h*)&As[p][r][16 * lh];       // K = 16*lh + 0..7
            u.h[1] = *(const v8h*)&As[p][r][16 * lh + 8];   // K = 16*lh + 8..15
            afr[mt] = u.v;
        }

        #pragma unroll
        for (int t = 0; t < 3; ++t) {
            const int nt = wave + t * 8;
            if (nt < NTILES) {
                if (kc + 1 < KCHUNKS)
                    __builtin_prefetch(wrow[t] + kb + 32, 0, 1);
                const v16h bfr = *(const v16h*)(wrow[t] + kb);  // 32B aligned load
                #pragma unroll
                for (int mt = 0; mt < 4; ++mt)
                    acc[mt][t] = __builtin_amdgcn_wmma_f32_16x16x32_f16(
                        false, afr[mt], false, bfr, (short)0, acc[mt][t], false, false);
            }
        }
    }

    // ---- epilogue: C layout (VGPR r: lanes0-15 M=r, lanes16-31 M=8+r) ----
    #pragma unroll
    for (int t = 0; t < 3; ++t) {
        const int nt = wave + t * 8;
        if (nt >= NTILES) continue;
        const int n = nt * 16 + lm;
        if (n >= Nstore) continue;
        const float badd = (bias && n < DD) ? bias[n] : 0.f;
        #pragma unroll
        for (int mt = 0; mt < 4; ++mt) {
            #pragma unroll
            for (int r = 0; r < 8; ++r) {
                const int m = mb + mt * 16 + r + 8 * lh;
                float v = acc[mt][t][r] + badd;
                if (res) v += res[(long)m * DD + n];
                C[(long)m * ldc + n] = v;
            }
        }
    }
}

// ---------------------------------------------------------------------------
// out[m] = dot(A[m, :300], v) + (*cadd)   -- one wave32 per row
// ---------------------------------------------------------------------------
__global__ __launch_bounds__(256)
void rowdot_kernel(const float* __restrict__ A, int lda, const float* __restrict__ v,
                   float* __restrict__ out, int M, const float* __restrict__ cadd)
{
    const int gw   = (blockIdx.x * 256 + threadIdx.x) >> 5;
    const int lane = threadIdx.x & 31;
    if (gw >= M) return;
    const float* row = A + (long)gw * lda;
    float s = 0.f;
    for (int k = lane; k < DD; k += 32) s += row[k] * v[k];
    #pragma unroll
    for (int off = 16; off; off >>= 1) s += __shfl_xor(s, off, 32);
    if (lane == 0) out[gw] = s + (cadd ? *cadd : 0.f);
}

// ---------------------------------------------------------------------------
// lens + position weights per (b,s)
// ---------------------------------------------------------------------------
__global__ __launch_bounds__(256)
void prep_wscale_kernel(const int* __restrict__ txt, float* __restrict__ wscale)
{
    const int b = blockIdx.x, tid = threadIdx.x;
    __shared__ int red[256];
    int c = 0;
    for (int s = tid; s < SS; s += 256) c += (txt[b * SS + s] != 0);
    red[tid] = c; __syncthreads();
    for (int o = 128; o; o >>= 1) { if (tid < o) red[tid] += red[tid + o]; __syncthreads(); }
    const int len = red[0];
    const float lf = (float)len;
    for (int s = tid; s < SS; s += 256) {
        const int j = s - (SS - len);
        wscale[b * SS + s] = (j >= 0) ? (1.f - (float)j / lf) : 1.f;
    }
}

// ---------------------------------------------------------------------------
// aspect[b,d] = mean of emb over nonzero aspect tokens (emb[0]==0)
// ---------------------------------------------------------------------------
__global__ __launch_bounds__(320)
void aspect_kernel(const int* __restrict__ asp, const float* __restrict__ emb,
                   float* __restrict__ aspect)
{
    const int b = blockIdx.x, d = threadIdx.x;
    int idx[8]; int n = 0;
    #pragma unroll
    for (int i = 0; i < 8; ++i) { idx[i] = asp[b * 8 + i]; n += (idx[i] != 0); }
    if (d < DD) {
        float s = 0.f;
        #pragma unroll
        for (int i = 0; i < 8; ++i) s += emb[(long)idx[i] * DD + d];
        aspect[b * DD + d] = s / (float)n;
    }
}

// ---------------------------------------------------------------------------
// u = Wq^T @ wq_part ; c0 = bq . wq_part   (folds one GEMM out of each hop)
// ---------------------------------------------------------------------------
__global__ __launch_bounds__(320)
void uvec_kernel(const float* __restrict__ Wq, const float* __restrict__ bq,
                 const float* __restrict__ wq, float* __restrict__ u,
                 float* __restrict__ c0)
{
    const int k = threadIdx.x;
    if (k < DD) {
        float s = 0.f;
        for (int n = 0; n < DD; ++n) s += Wq[n * DD + k] * wq[n];
        u[k] = s;
    }
    if (k == 0) {
        float s = 0.f;
        for (int n = 0; n < DD; ++n) s += bq[n] * wq[n];
        *c0 = s;
    }
}

// ---------------------------------------------------------------------------
// per-batch: score=tanh(k_score+qdot); softmax over all S; out = attn @ kx
// ---------------------------------------------------------------------------
__global__ __launch_bounds__(512)
void attn_kernel(const float* __restrict__ k_score, const float* __restrict__ qdot,
                 const float* __restrict__ kx, float* __restrict__ out)
{
    const int b = blockIdx.x, tid = threadIdx.x;
    __shared__ float attn[SS];
    __shared__ float red[SS];
    const float scv = tanhf(k_score[b * SS + tid] + qdot[b]);
    red[tid] = scv; __syncthreads();
    for (int o = 256; o; o >>= 1) { if (tid < o) red[tid] = fmaxf(red[tid], red[tid + o]); __syncthreads(); }
    const float mx = red[0]; __syncthreads();
    const float e = expf(scv - mx);
    red[tid] = e; __syncthreads();
    for (int o = 256; o; o >>= 1) { if (tid < o) red[tid] += red[tid + o]; __syncthreads(); }
    attn[tid] = e / red[0];
    __syncthreads();
    if (tid < DD) {
        float acc = 0.f;
        const float* kbp = kx + (long)b * SS * LDKX + tid;
        #pragma unroll 4
        for (int s = 0; s < SS; ++s) acc += attn[s] * kbp[(long)s * LDKX];
        out[b * DD + tid] = acc;
    }
}

// ---------------------------------------------------------------------------
// final: out[b,p] = x[b,:].Wd[p,:] + bd[p]   -- 3 waves per batch row
// ---------------------------------------------------------------------------
__global__ __launch_bounds__(96)
void final_kernel(const float* __restrict__ x, const float* __restrict__ Wd,
                  const float* __restrict__ bd, float* __restrict__ out)
{
    const int b = blockIdx.x;
    const int p = threadIdx.x >> 5, lane = threadIdx.x & 31;
    float s = 0.f;
    for (int k = lane; k < DD; k += 32) s += x[b * DD + k] * Wd[p * DD + k];
    #pragma unroll
    for (int off = 16; off; off >>= 1) s += __shfl_xor(s, off, 32);
    if (lane == 0) out[b * PP + p] = s + bd[p];
}

// ---------------------------------------------------------------------------
extern "C" void kernel_launch(void* const* d_in, const int* in_sizes, int n_in,
                              void* d_out, int out_size, void* d_ws, size_t ws_size,
                              hipStream_t stream)
{
    const int*   txt  = (const int*)  d_in[0];
    const int*   asp  = (const int*)  d_in[1];
    const float* emb  = (const float*)d_in[2];
    const float* Wx   = (const float*)d_in[3];
    const float* bx   = (const float*)d_in[4];
    const float* Wk   = (const float*)d_in[5];
    const float* bk   = (const float*)d_in[6];
    const float* Wq   = (const float*)d_in[7];
    const float* bq   = (const float*)d_in[8];
    const float* wmlp = (const float*)d_in[9];
    const float* Wp   = (const float*)d_in[10];
    const float* bp   = (const float*)d_in[11];
    const float* Wd   = (const float*)d_in[12];
    const float* bd   = (const float*)d_in[13];
    float* outp = (float*)d_out;

    // workspace carve-up (256B aligned)
    char* base = (char*)d_ws;
    size_t off = 0;
    auto carve = [&](size_t bytes) -> void* {
        void* p = (void*)(base + off);
        off += (bytes + 255) & ~(size_t)255;
        return p;
    };
    float*    kx      = (float*)   carve((size_t)BB * SS * LDKX * 4);   // ~319 MB
    float*    wscale  = (float*)   carve((size_t)BB * SS * 4);
    float*    kscore  = (float*)   carve((size_t)BB * SS * 4);
    float*    aspectb = (float*)   carve((size_t)BB * DD * 4);
    float*    xb0     = (float*)   carve((size_t)BB * DD * 4);
    float*    xb1     = (float*)   carve((size_t)BB * DD * 4);
    float*    xb2     = (float*)   carve((size_t)BB * DD * 4);
    float*    attout  = (float*)   carve((size_t)BB * DD * 4);
    float*    qdot    = (float*)   carve((size_t)BB * 4);
    float*    uvec    = (float*)   carve((size_t)DD * 4);
    float*    c0      = (float*)   carve(4);
    _Float16* whk     = (_Float16*)carve((size_t)WH_ROWS * WH_LDN * 2);
    _Float16* whx     = (_Float16*)carve((size_t)WH_ROWS * WH_LDN * 2);
    _Float16* whp     = (_Float16*)carve((size_t)WH_ROWS * WH_LDN * 2);
    (void)ws_size; (void)in_sizes; (void)n_in; (void)out_size;

    // ---- prep ----
    const int cgrid = (WH_ROWS * WH_LDN + 255) / 256;
    convert_w_kernel<<<cgrid, 256, 0, stream>>>(Wk, whk);
    convert_w_kernel<<<cgrid, 256, 0, stream>>>(Wx, whx);
    convert_w_kernel<<<cgrid, 256, 0, stream>>>(Wp, whp);
    prep_wscale_kernel<<<BB, 256, 0, stream>>>(txt, wscale);
    aspect_kernel<<<BB, 320, 0, stream>>>(asp, emb, aspectb);
    uvec_kernel<<<1, 320, 0, stream>>>(Wq, bq, wmlp + DD, uvec, c0);

    // ---- kx = (w * emb[text]) @ Wk^T + bk  (the big WMMA GEMM) ----
    const int Mbig = BB * SS;
    wmma_gemm_nt<<<Mbig / 64, 256, 0, stream>>>(
        nullptr, txt, emb, wscale, whk, bk, nullptr, kx, LDKX, Mbig, LDKX);

    // ---- k_score = kx . w_mlp[:300] ----
    rowdot_kernel<<<Mbig / 8, 256, 0, stream>>>(kx, LDKX, wmlp, kscore, Mbig, nullptr);

    // ---- hops (explicit buffer plan: xh/xnew never alias xin) ----
    const float* xin_tab[3]  = {aspectb, xb1, xb0};
    float*       xh_tab[3]   = {xb0,     xb2, xb1};
    float*       xnew_tab[3] = {xb1,     xb0, xb2};
    for (int hop = 0; hop < 3; ++hop) {
        const float* xin  = xin_tab[hop];
        float*       xh   = xh_tab[hop];
        float*       xnew = xnew_tab[hop];
        // x1 = xin @ Wx^T + bx
        wmma_gemm_nt<<<BB / 64, 256, 0, stream>>>(
            xin, nullptr, nullptr, nullptr, whx, bx, nullptr, xh, DD, BB, DD);
        // qdot[b] = x1 . u + c0   (== (x1@Wq^T+bq).wq_part)
        rowdot_kernel<<<BB / 8, 256, 0, stream>>>(xh, DD, uvec, qdot, BB, c0);
        // softmax + weighted sum over kx
        attn_kernel<<<BB, 512, 0, stream>>>(kscore, qdot, kx, attout);
        // xnew = attout @ Wp^T + bp + x1
        wmma_gemm_nt<<<BB / 64, 256, 0, stream>>>(
            attout, nullptr, nullptr, nullptr, whp, bp, xh, xnew, DD, BB, DD);
    }

    // ---- result = x @ Wd^T + bd ----
    final_kernel<<<BB, 96, 0, stream>>>(xb2, Wd, bd, outp);
}
